// MockQwenForCausalLM_3624952398524
// MI455X (gfx1250) — matrix operand order repaired and checked by reading, not compiled, not run on hardware
//
#include <hip/hip_runtime.h>

#define HIDDEN 2048
#define VOCAB  32000
#define BATCH  4
#define SEQ    2048
#define IGNORE_INDEX (-100)

typedef __attribute__((ext_vector_type(16))) __bf16 bf16x16;
typedef __attribute__((ext_vector_type(8)))  __bf16 bf16x8;
typedef __attribute__((ext_vector_type(8)))  float  f32x8;

#ifndef __has_builtin
#define __has_builtin(x) 0
#endif

#if __has_builtin(__builtin_amdgcn_global_load_async_to_lds_b128)
#define HAVE_ASYNC_LDS 1
#else
#define HAVE_ASYNC_LDS 0
#endif

#if HAVE_ASYNC_LDS
#if __has_builtin(__builtin_amdgcn_s_wait_asynccnt)
#define WAIT_ASYNC(n) __builtin_amdgcn_s_wait_asynccnt(n)
#else
#define WAIT_ASYNC(n) asm volatile("s_wait_asynccnt %0" ::"n"(n) : "memory")
#endif
// one 16B per-lane async copy: global -> LDS, tracked by ASYNCcnt
typedef int vi4_t __attribute__((vector_size(16)));
__device__ __forceinline__ void async_cp16(const __bf16* g, __bf16* l) {
    __builtin_amdgcn_global_load_async_to_lds_b128(
        (__attribute__((address_space(1))) vi4_t*)g,
        (__attribute__((address_space(3))) vi4_t*)l, 0, 0);
}
#endif

// ---------------------------------------------------------------------------
// Kernel: zero the loss accumulators in workspace
// ---------------------------------------------------------------------------
__global__ void init_accum_kernel(float* lossAcc, int* cntAcc) {
    *lossAcc = 0.0f;
    *cntAcc  = 0;
}

// ---------------------------------------------------------------------------
// Kernel: convert W (VOCAB x HIDDEN fp32) -> bf16, vectorized 4-wide
// ---------------------------------------------------------------------------
__global__ __launch_bounds__(256)
void convert_w_bf16_kernel(const float* __restrict__ W, __bf16* __restrict__ Wb,
                           long long n4) {
    long long i = (long long)blockIdx.x * blockDim.x + threadIdx.x;
    if (i >= n4) return;
    float4 v = ((const float4*)W)[i];
    union { __bf16 h[4]; uint2 u; } t;
    t.h[0] = (__bf16)v.x; t.h[1] = (__bf16)v.y;
    t.h[2] = (__bf16)v.z; t.h[3] = (__bf16)v.w;
    ((uint2*)Wb)[i] = t.u;
}

// ---------------------------------------------------------------------------
// Kernel: embedding gather + fp32->bf16  (one block per token row)
// ---------------------------------------------------------------------------
__global__ __launch_bounds__(256)
void gather_embed_bf16_kernel(const int* __restrict__ ids,
                              const float* __restrict__ embed,
                              __bf16* __restrict__ Ab) {
    const int row = blockIdx.x;                 // 0 .. BATCH*SEQ-1
    const int tok = ids[row];
    const float4* src = (const float4*)(embed + (long long)tok * HIDDEN);
    uint2* dst = (uint2*)(Ab + (long long)row * HIDDEN);
    for (int j = threadIdx.x; j < HIDDEN / 4; j += 256) {
        float4 v = src[j];
        union { __bf16 h[4]; uint2 u; } t;
        t.h[0] = (__bf16)v.x; t.h[1] = (__bf16)v.y;
        t.h[2] = (__bf16)v.z; t.h[3] = (__bf16)v.w;
        dst[j] = t.u;
    }
}

// ---------------------------------------------------------------------------
// Core GEMM:  C[M,N] = A[M,K] * B[N,K]^T   (A,B bf16, C fp32)
// M=8192, N=32000, K=2048.  Block tile 128x128, K-tile 32.
// 8 waves (wave32) per block, 4x2 wave grid, each wave = 2x4 WMMA tiles.
// LDS double-buffered; tiles staged with GLOBAL_LOAD_ASYNC_TO_LDS_B128
// (ASYNCcnt) when the toolchain exposes it, else reg double-buffer fallback.
// ---------------------------------------------------------------------------
#define BM 128
#define BN 128
#define BK 32
#define LDT 40                 // padded LDS row stride (80B) to stagger banks
#define KTILES (HIDDEN / BK)
#define TILE_ELEMS (BM * LDT)  // 5120 bf16 per tile buffer

__global__ __launch_bounds__(256)
void gemm_bf16_wmma_kernel(const __bf16* __restrict__ A,
                           const __bf16* __restrict__ B,
                           float* __restrict__ C) {
    __shared__ __bf16 As[2 * TILE_ELEMS];
    __shared__ __bf16 Bs[2 * TILE_ELEMS];

    const int tid  = threadIdx.x;
    const int lane = tid & 31;
    const int wave = tid >> 5;
    const int wm   = wave & 3;     // 0..3 -> 32 M-rows each
    const int wn   = wave >> 2;    // 0..1 -> 64 N-cols each

    const long long tileM = (long long)blockIdx.y * BM;
    const long long tileN = (long long)blockIdx.x * BN;

    // --- staging: 512 chunks of 16B per tile, 2 chunks per thread -----------
    const int c0 = tid, c1 = tid + 256;
    const int rA0 = c0 >> 2, oA0 = (c0 & 3) * 8;   // row in tile, elem offset
    const int rA1 = c1 >> 2, oA1 = (c1 & 3) * 8;

    const __bf16* gA0 = A + (tileM + rA0) * HIDDEN + oA0;
    const __bf16* gA1 = A + (tileM + rA1) * HIDDEN + oA1;
    const __bf16* gB0 = B + (tileN + rA0) * HIDDEN + oA0;
    const __bf16* gB1 = B + (tileN + rA1) * HIDDEN + oA1;

    __bf16* lA0 = &As[rA0 * LDT + oA0];
    __bf16* lA1 = &As[rA1 * LDT + oA1];
    __bf16* lB0 = &Bs[rA0 * LDT + oA0];
    __bf16* lB1 = &Bs[rA1 * LDT + oA1];

    // --- fragment element indices (ISA 7.12.2 16-bit layouts) ---------------
    const int hrow  = lane & 15;
    const int kOffA = (lane & 16) ? 8  : 0;  // A: half-wave K split {0..7,16..23}
    const int kOffB = (lane & 16) ? 8  : 0;  // B: half-wave K split {0..7,16..23}
    int aIdx[2], bIdx[4];
#pragma unroll
    for (int i = 0; i < 2; ++i)
        aIdx[i] = (wm * 32 + i * 16 + hrow) * LDT + kOffA;
#pragma unroll
    for (int j = 0; j < 4; ++j)
        bIdx[j] = (wn * 64 + j * 16 + hrow) * LDT + kOffB;

    f32x8 acc[2][4] = {};

    auto compute_tile = [&](int bufOff) {
        bf16x16 af[2];
#pragma unroll
        for (int i = 0; i < 2; ++i) {
            const __bf16* p = &As[bufOff + aIdx[i]];
            bf16x8 lo = *(const bf16x8*)(p);
            bf16x8 hi = *(const bf16x8*)(p + 16);
            af[i] = __builtin_shufflevector(lo, hi, 0,1,2,3,4,5,6,7,
                                                     8,9,10,11,12,13,14,15);
        }
        bf16x16 bfm[4];
#pragma unroll
        for (int j = 0; j < 4; ++j) {
            const __bf16* p = &Bs[bufOff + bIdx[j]];
            bf16x8 lo = *(const bf16x8*)(p);
            bf16x8 hi = *(const bf16x8*)(p + 16);
            bfm[j] = __builtin_shufflevector(lo, hi, 0,1,2,3,4,5,6,7,
                                                      8,9,10,11,12,13,14,15);
        }
#pragma unroll
        for (int i = 0; i < 2; ++i)
#pragma unroll
            for (int j = 0; j < 4; ++j)
                acc[i][j] = __builtin_amdgcn_wmma_f32_16x16x32_bf16(
                    false, af[i], false, bfm[j], (short)0, acc[i][j],
                    false, false);
    };

#if HAVE_ASYNC_LDS
    // ---------------- async-to-LDS pipeline, LDS double buffered ------------
    async_cp16(gA0, lA0); async_cp16(gA1, lA1);
    async_cp16(gB0, lB0); async_cp16(gB1, lB1);

    for (int kt = 0; kt < KTILES; ++kt) {
        WAIT_ASYNC(0);                 // my async fills of buffer `cur` done
        __syncthreads();               // everyone done filling + done reading nxt
        const int cur = (kt & 1) * TILE_ELEMS;
        if (kt + 1 < KTILES) {
            const int ko  = (kt + 1) * BK;
            const int nxt = ((kt + 1) & 1) * TILE_ELEMS;
            async_cp16(gA0 + ko, lA0 + nxt);
            async_cp16(gA1 + ko, lA1 + nxt);
            async_cp16(gB0 + ko, lB0 + nxt);
            async_cp16(gB1 + ko, lB1 + nxt);
        }
        if (kt + 2 < KTILES) {         // pull K-tiles ahead into GL2
            __builtin_prefetch(gA0 + (kt + 2) * BK, 0, 2);
            __builtin_prefetch(gB0 + (kt + 2) * BK, 0, 2);
        }
        compute_tile(cur);
    }
#else
    // ---------------- fallback: register double-buffer, single LDS buffer ---
    uint4 ra0 = *(const uint4*)(gA0);
    uint4 ra1 = *(const uint4*)(gA1);
    uint4 rb0 = *(const uint4*)(gB0);
    uint4 rb1 = *(const uint4*)(gB1);

    for (int kt = 0; kt < KTILES; ++kt) {
        __syncthreads();
        *(uint4*)lA0 = ra0; *(uint4*)lA1 = ra1;
        *(uint4*)lB0 = rb0; *(uint4*)lB1 = rb1;
        __syncthreads();
        if (kt + 1 < KTILES) {
            const int ko = (kt + 1) * BK;
            ra0 = *(const uint4*)(gA0 + ko);
            ra1 = *(const uint4*)(gA1 + ko);
            rb0 = *(const uint4*)(gB0 + ko);
            rb1 = *(const uint4*)(gB1 + ko);
        }
        if (kt + 2 < KTILES) {
            __builtin_prefetch(gA0 + (kt + 2) * BK, 0, 2);
            __builtin_prefetch(gB0 + (kt + 2) * BK, 0, 2);
        }
        compute_tile(0);
    }
#endif

    // --- epilogue: C/D layout — VGPR r: M=r (lanes 0-15) / M=8+r (16-31) ----
    const int rbase = (lane & 16) ? 8 : 0;
    const int col   = lane & 15;
#pragma unroll
    for (int i = 0; i < 2; ++i) {
#pragma unroll
        for (int j = 0; j < 4; ++j) {
            const long long gn = tileN + wn * 64 + j * 16 + col;
#pragma unroll
            for (int r = 0; r < 8; ++r) {
                const long long gm = tileM + wm * 32 + i * 16 + rbase + r;
                C[gm * VOCAB + gn] = acc[i][j][r];
            }
        }
    }
}

// ---------------------------------------------------------------------------
// Cross-entropy: one block per shifted row (B*(S-1) rows)
// ---------------------------------------------------------------------------
__global__ __launch_bounds__(256)
void ce_rows_kernel(const float* __restrict__ logits,
                    const int* __restrict__ labels,
                    float* __restrict__ lossAcc, int* __restrict__ cntAcc) {
    __shared__ float sdata[256];
    const int blk = blockIdx.x;
    const int b = blk / (SEQ - 1);
    const int s = blk % (SEQ - 1);
    const float* row = logits + (long long)(b * SEQ + s) * VOCAB;
    const int label = labels[b * SEQ + s + 1];

    float m = -3.4e38f;
    for (int j = threadIdx.x; j < VOCAB; j += 256) m = fmaxf(m, row[j]);
    sdata[threadIdx.x] = m;
    __syncthreads();
    for (int o = 128; o > 0; o >>= 1) {
        if (threadIdx.x < o)
            sdata[threadIdx.x] = fmaxf(sdata[threadIdx.x], sdata[threadIdx.x + o]);
        __syncthreads();
    }
    const float rowmax = sdata[0];
    __syncthreads();

    float sum = 0.0f;
    for (int j = threadIdx.x; j < VOCAB; j += 256) sum += __expf(row[j] - rowmax);
    sdata[threadIdx.x] = sum;
    __syncthreads();
    for (int o = 128; o > 0; o >>= 1) {
        if (threadIdx.x < o) sdata[threadIdx.x] += sdata[threadIdx.x + o];
        __syncthreads();
    }

    if (threadIdx.x == 0 && label != IGNORE_INDEX) {
        float nll = -(row[label] - rowmax - __logf(sdata[0]));
        atomicAdd(lossAcc, nll);
        atomicAdd(cntAcc, 1);
    }
}

__global__ void finalize_loss_kernel(const float* lossAcc, const int* cntAcc,
                                     float* out) {
    int c = *cntAcc;
    out[0] = *lossAcc / (float)(c > 0 ? c : 1);
}

// ---------------------------------------------------------------------------
// Launch
// ---------------------------------------------------------------------------
extern "C" void kernel_launch(void* const* d_in, const int* in_sizes, int n_in,
                              void* d_out, int out_size, void* d_ws, size_t ws_size,
                              hipStream_t stream) {
    const int*   ids    = (const int*)d_in[0];
    const int*   labels = (const int*)d_in[1];
    const float* embed  = (const float*)d_in[2];
    const float* W      = (const float*)d_in[3];

    float* out    = (float*)d_out;
    float* logits = out + 1;                    // tuple: (loss, logits)

    __bf16* Ab = (__bf16*)d_ws;                              // [B*S, H] bf16
    __bf16* Wb = Ab + (size_t)BATCH * SEQ * HIDDEN;          // [V, H]  bf16
    float*  lossAcc = (float*)(Wb + (size_t)VOCAB * HIDDEN);
    int*    cntAcc  = (int*)(lossAcc + 1);

    init_accum_kernel<<<1, 1, 0, stream>>>(lossAcc, cntAcc);

    const long long n4 = (long long)VOCAB * HIDDEN / 4;
    convert_w_bf16_kernel<<<(int)((n4 + 255) / 256), 256, 0, stream>>>(W, Wb, n4);

    gather_embed_bf16_kernel<<<BATCH * SEQ, 256, 0, stream>>>(ids, embed, Ab);

    dim3 grid(VOCAB / BN, (BATCH * SEQ) / BM);   // 250 x 64
    gemm_bf16_wmma_kernel<<<grid, 256, 0, stream>>>(Ab, Wb, logits);

    ce_rows_kernel<<<BATCH * (SEQ - 1), 256, 0, stream>>>(logits, labels,
                                                          lossAcc, cntAcc);
    finalize_loss_kernel<<<1, 1, 0, stream>>>(lossAcc, cntAcc, out);
}
